// AFTSimple_87170656240173
// MI455X (gfx1250) — compile-verified, matrix operand order checked by reading, not run on hardware
//
#include <hip/hip_runtime.h>
#include <hip/hip_bf16.h>
#include <math.h>

// ---------------------------------------------------------------------------
// Problem constants (B=1, S=8192, E=1024)
// ---------------------------------------------------------------------------
#define S_DIM 8192
#define E_DIM 1024
#define TM 128          // workgroup M tile
#define TN 128          // workgroup N tile
#define TK 32           // K step (one bf16 WMMA K)
#define NTHREADS 256    // 8 waves (wave32)
#define LDS_STRIDE 40   // ushorts per row (32 + 8 pad) -> 80B, 16B aligned, bank-friendly
#define CHUNKS 64       // row chunks for column reductions (8192/128)

typedef __attribute__((ext_vector_type(16))) __bf16 v16bf;
typedef __attribute__((ext_vector_type(8)))  float  v8f;

// RNE float -> bf16 bits
__device__ __forceinline__ unsigned short f2bf(float f) {
    unsigned u = __float_as_uint(f);
    unsigned r = u + 0x7FFFu + ((u >> 16) & 1u);
    return (unsigned short)(r >> 16);
}
__device__ __forceinline__ float bf2f(unsigned short h) {
    return __uint_as_float(((unsigned)h) << 16);
}

// Load a 16-lane WMMA operand fragment from LDS.
// Per-lane layout (both A 16x32 and B 32x16 use the same gather on CDNA5):
//   row  = lane & 15 (M for A, N for B)
//   K runs: [8h .. 8h+7] and [16+8h .. 16+8h+7], h = lane>>4
__device__ __forceinline__ v16bf load_frag(const unsigned short* arr, int row, int halfK) {
    const unsigned short* base = arr + row * LDS_STRIDE;
    uint4 x = *(const uint4*)(base + 8 * halfK);        // K = 8h..8h+7
    uint4 y = *(const uint4*)(base + 16 + 8 * halfK);   // K = 16+8h..16+8h+7
    struct alignas(32) Pack { uint4 a, b; } pk{ x, y };
    v16bf v;
    __builtin_memcpy(&v, &pk, sizeof(v));
    return v;
}

__device__ __forceinline__ v8f wmma_bf16(v16bf a, v16bf b, v8f c) {
    return __builtin_amdgcn_wmma_f32_16x16x32_bf16(
        /*neg_a=*/false, a, /*neg_b=*/false, b,
        /*c_mod=*/(short)0, c, /*reuse_a=*/false, /*reuse_b=*/false);
}

// ---------------------------------------------------------------------------
// C[s,f] = sum_e A[s,e] * W[f,e] + bias[f]       (both operands K-contiguous)
// mode 0: store C          mode 1: store wc * sigmoid(C)   (wc = *wcPtr)
// bf16 hi/lo split: a*b ~= aHi*bHi + aHi*bLo + aLo*bHi  (fp32-class accuracy)
// ---------------------------------------------------------------------------
__global__ __launch_bounds__(NTHREADS) void gemm_split_bf16(
    const float* __restrict__ A, const float* __restrict__ W,
    const float* __restrict__ bias, float* __restrict__ C,
    const float* __restrict__ wcPtr, int mode)
{
    __shared__ alignas(16) unsigned short sAhi[TM * LDS_STRIDE];
    __shared__ alignas(16) unsigned short sAlo[TM * LDS_STRIDE];
    __shared__ alignas(16) unsigned short sBhi[TN * LDS_STRIDE];
    __shared__ alignas(16) unsigned short sBlo[TN * LDS_STRIDE];

    const int t     = threadIdx.x;
    const int wave  = t >> 5;
    const int lane  = t & 31;
    const int row16 = lane & 15;
    const int halfK = lane >> 4;
    const int tileM = blockIdx.x * TM;
    const int tileN = blockIdx.y * TN;

    v8f zero = {0.f, 0.f, 0.f, 0.f, 0.f, 0.f, 0.f, 0.f};
    v8f acc[8];
#pragma unroll
    for (int n = 0; n < 8; ++n) acc[n] = zero;

    for (int kk = 0; kk < E_DIM; kk += TK) {
        // ---- stage fp32 tiles -> bf16 hi/lo in LDS (4 float4 per thread per matrix)
#pragma unroll
        for (int i = 0; i < 4; ++i) {
            int idx = t + i * NTHREADS;            // 0..1023 float4 slots
            int r   = idx >> 3;                    // tile row 0..127
            int c4  = idx & 7;                     // float4 within the 32-K row
            float4 a4 = *(const float4*)(A + (size_t)(tileM + r) * E_DIM + kk + c4 * 4);
            float4 b4 = *(const float4*)(W + (size_t)(tileN + r) * E_DIM + kk + c4 * 4);
            const float av[4] = {a4.x, a4.y, a4.z, a4.w};
            const float bv[4] = {b4.x, b4.y, b4.z, b4.w};
#pragma unroll
            for (int j = 0; j < 4; ++j) {
                unsigned short ah = f2bf(av[j]);
                unsigned short bh = f2bf(bv[j]);
                sAhi[r * LDS_STRIDE + c4 * 4 + j] = ah;
                sBhi[r * LDS_STRIDE + c4 * 4 + j] = bh;
                sAlo[r * LDS_STRIDE + c4 * 4 + j] = f2bf(av[j] - bf2f(ah));
                sBlo[r * LDS_STRIDE + c4 * 4 + j] = f2bf(bv[j] - bf2f(bh));
            }
        }
        __syncthreads();

        // prefetch next K-stage (global_prefetch_b8), one row per thread-pair
        if (kk + TK < E_DIM) {
            int pr = t & (TM - 1);
            __builtin_prefetch(A + (size_t)(tileM + pr) * E_DIM + kk + TK, 0, 0);
            __builtin_prefetch(W + (size_t)(tileN + pr) * E_DIM + kk + TK, 0, 0);
        }

        // ---- compute: 8 N-subtiles x 3 split terms = 24 WMMA per K-stage
        v16bf aHi = load_frag(sAhi, wave * 16 + row16, halfK);
        v16bf aLo = load_frag(sAlo, wave * 16 + row16, halfK);
#pragma unroll
        for (int n = 0; n < 8; ++n) {
            v16bf bHi = load_frag(sBhi, n * 16 + row16, halfK);
            v16bf bLo = load_frag(sBlo, n * 16 + row16, halfK);
            acc[n] = wmma_bf16(aHi, bHi, acc[n]);
            acc[n] = wmma_bf16(aHi, bLo, acc[n]);
            acc[n] = wmma_bf16(aLo, bHi, acc[n]);
        }
        __syncthreads();
    }

    // ---- epilogue: C/D layout -> lane row = r + 8*halfK, col = n*16 + (lane&15)
    float wc = (mode == 1) ? *wcPtr : 0.f;
#pragma unroll
    for (int n = 0; n < 8; ++n) {
        int col  = tileN + n * 16 + row16;
        float bb = bias[col];
#pragma unroll
        for (int r = 0; r < 8; ++r) {
            int row = tileM + wave * 16 + r + 8 * halfK;
            float v = acc[n][r] + bb;
            if (mode == 1) v = wc / (1.f + expf(-v));   // wc * sigmoid(v)
            C[(size_t)row * E_DIM + col] = v;
        }
    }
}

// ---------------------------------------------------------------------------
// Column softmax reduction chain (softmax over the sequence axis per feature)
// ---------------------------------------------------------------------------
__global__ void colmax_partial(const float* __restrict__ K, float* __restrict__ pmax) {
    int f = blockIdx.x * 256 + threadIdx.x;
    int c = blockIdx.y;
    float m = -3.402823466e38f;
    for (int s = c * 128; s < c * 128 + 128; ++s)
        m = fmaxf(m, K[(size_t)s * E_DIM + f]);
    pmax[c * E_DIM + f] = m;
}

__global__ void colmax_final(const float* __restrict__ pmax, float* __restrict__ colmax) {
    int f = blockIdx.x * 256 + threadIdx.x;
    float m = -3.402823466e38f;
    for (int c = 0; c < CHUNKS; ++c) m = fmaxf(m, pmax[c * E_DIM + f]);
    colmax[f] = m;
}

__global__ void colsum_partial(const float* __restrict__ K, const float* __restrict__ V,
                               const float* __restrict__ colmax,
                               float* __restrict__ pden, float* __restrict__ pnum) {
    int f = blockIdx.x * 256 + threadIdx.x;
    int c = blockIdx.y;
    float m = colmax[f];
    float den = 0.f, num = 0.f;
    for (int s = c * 128; s < c * 128 + 128; ++s) {
        float e = expf(K[(size_t)s * E_DIM + f] - m);
        den += e;
        num += e * V[(size_t)s * E_DIM + f];
    }
    pden[c * E_DIM + f] = den;
    pnum[c * E_DIM + f] = num;
}

__global__ __launch_bounds__(1024) void finalize_wc(const float* __restrict__ pden,
                                                    const float* __restrict__ pnum,
                                                    float* __restrict__ wc) {
    __shared__ float red[1024];
    int f = threadIdx.x;
    float den = 0.f, num = 0.f;
    for (int c = 0; c < CHUNKS; ++c) {
        den += pden[c * E_DIM + f];
        num += pnum[c * E_DIM + f];
    }
    red[f] = num / den;
    __syncthreads();
    for (int st = 512; st > 0; st >>= 1) {
        if (f < st) red[f] += red[f + st];
        __syncthreads();
    }
    if (f == 0) *wc = red[0];
}

// ---------------------------------------------------------------------------
extern "C" void kernel_launch(void* const* d_in, const int* in_sizes, int n_in,
                              void* d_out, int out_size, void* d_ws, size_t ws_size,
                              hipStream_t stream) {
    (void)in_sizes; (void)n_in; (void)out_size; (void)ws_size;
    const float* q  = (const float*)d_in[0];
    const float* Wq = (const float*)d_in[1];
    const float* bq = (const float*)d_in[2];
    const float* Wk = (const float*)d_in[3];
    const float* bk = (const float*)d_in[4];
    const float* Wv = (const float*)d_in[5];
    const float* bv = (const float*)d_in[6];
    float* out = (float*)d_out;

    const size_t SE = (size_t)S_DIM * E_DIM;
    float* ws      = (float*)d_ws;
    float* Kbuf    = ws;                       // [S,E]
    float* Vbuf    = ws + SE;                  // [S,E]
    float* pmax    = ws + 2 * SE;              // [CHUNKS,E]
    float* pden    = pmax + CHUNKS * E_DIM;    // [CHUNKS,E]
    float* pnum    = pden + CHUNKS * E_DIM;    // [CHUNKS,E]
    float* colmax  = pnum + CHUNKS * E_DIM;    // [E]
    float* wcBuf   = colmax + E_DIM;           // [1]

    dim3 gGemm(S_DIM / TM, E_DIM / TN);        // 64 x 8
    dim3 bGemm(NTHREADS);
    dim3 gRed(E_DIM / 256, CHUNKS);            // 4 x 64

    // K = q @ Wk^T + bk ; V = q @ Wv^T + bv   (fp32 to workspace)
    gemm_split_bf16<<<gGemm, bGemm, 0, stream>>>(q, Wk, bk, Kbuf, nullptr, 0);
    gemm_split_bf16<<<gGemm, bGemm, 0, stream>>>(q, Wv, bv, Vbuf, nullptr, 0);

    // wc = sum_f ( sum_s exp(K-m_f) * V ) / ( sum_s exp(K-m_f) )
    colmax_partial<<<gRed, 256, 0, stream>>>(Kbuf, pmax);
    colmax_final<<<E_DIM / 256, 256, 0, stream>>>(pmax, colmax);
    colsum_partial<<<gRed, 256, 0, stream>>>(Kbuf, Vbuf, colmax, pden, pnum);
    finalize_wc<<<1, 1024, 0, stream>>>(pden, pnum, wcBuf);

    // Y = sigmoid(q @ Wq^T + bq) * wc    (fused epilogue, straight to d_out)
    gemm_split_bf16<<<gGemm, bGemm, 0, stream>>>(q, Wq, bq, out, wcBuf, 1);
}
